// RenderNet_L_39092792328242
// MI455X (gfx1250) — matrix operand order repaired and checked by reading, not compiled, not run on hardware
//
#include <hip/hip_runtime.h>

// ---------------------------------------------------------------------------
// SIREN 1x1-conv MLP, fused, for gfx1250 (MI455X).
//   out = z + 0.5*sin(30*(W4 @ sin(30*(W2 @ sin(30*(W1 @ x + b1)) + b2)) + b4))
//   x = concat(s, n, pixel_coords, view_direction) : 12 channels, N = 131072.
// Strategy: f16 WMMA (16x16x32) with f32 accumulate; W2 A-fragments pinned in
// registers per wave; h1/h2 staged in LDS; GEMM3 via per-wave partial WMMA +
// LDS reduction.
// ---------------------------------------------------------------------------

typedef _Float16 h16;
typedef __attribute__((ext_vector_type(16))) _Float16 v16h;
typedef __attribute__((ext_vector_type(8)))  _Float16 v8h;
typedef __attribute__((ext_vector_type(8)))  float    v8f;

#define NPOS       131072        // 8*128*128 spatial positions
#define OMEGA      30.0f
#define LDS_STRIDE 264           // 256 K-halves + 8 pad (keeps 16B alignment)
#define NWG        1024
#define ITERS      4             // 1024 wg * 4 iters * 32 cols = 131072
#define NT         32            // columns per iteration (2 x 16-wide cblocks)

// A-fragment element mapping (16-bit A 16x32, contiguous k order used here, B matches):
//   lane l: M = l & 15, group g = l >> 4
//   half  h: K = (g << 4) + h
__device__ __forceinline__ int afrag_k(int lane, int h) {
    return ((lane >> 4) << 4) + h;
}

// ---------------------------------------------------------------------------
// Prologue: convert W1/W2/W4 (f32) into f16 A-fragments in d_ws.
//   tiles:  [0,16)   A1 row-blocks (K padded 12->32 with zeros)
//           [16,144) A2 row-block x k-tile (rb*8 + t)
//           [144,152) A3 k-tiles (rows 3..15 zeroed)
//   layout: wsA[tile*512 + lane*16 + h]
// ---------------------------------------------------------------------------
__global__ __launch_bounds__(256) void mlp_prep(const float* __restrict__ W1,
                                                const float* __restrict__ W2,
                                                const float* __restrict__ W4,
                                                h16* __restrict__ wsA) {
    int idx = blockIdx.x * 256 + threadIdx.x;       // < 152*512 = 77824
    int tile = idx >> 9;
    int rsub = idx & 511;
    int lane = rsub >> 4;
    int h    = rsub & 15;
    int K    = afrag_k(lane, h);
    int M    = lane & 15;
    float v;
    if (tile < 16) {                                 // W1 : [256][12]
        int rb = tile;
        v = (K < 12) ? W1[(rb * 16 + M) * 12 + K] : 0.0f;
    } else if (tile < 144) {                         // W2 : [256][256]
        int u = tile - 16, rb = u >> 3, t = u & 7;
        v = W2[(rb * 16 + M) * 256 + t * 32 + K];
    } else {                                         // W4 : [3][256]
        int t = tile - 144;
        v = (M < 3) ? W4[M * 256 + t * 32 + K] : 0.0f;
    }
    wsA[idx] = (h16)v;
}

// ---------------------------------------------------------------------------
// Main fused MLP kernel. 256 threads = 8 waves. Wave w owns output channel
// rows [32w, 32w+32) (2 row-blocks) and k-tile w of GEMM3.
// ---------------------------------------------------------------------------
__global__ __launch_bounds__(256) void mlp_main(const float* __restrict__ z,
                                                const float* __restrict__ nrm,
                                                const float* __restrict__ s,
                                                const float* __restrict__ pix,
                                                const float* __restrict__ vdir,
                                                const h16*  __restrict__ wsA,
                                                const float* __restrict__ b1,
                                                const float* __restrict__ b2,
                                                const float* __restrict__ b4,
                                                float* __restrict__ out) {
    // [which(h1/h2)][col][K]; partials for GEMM3 alias Hbuf[0] once h1 is dead.
    __shared__ __align__(16) h16 Hbuf[2][NT][LDS_STRIDE];
    float* part = reinterpret_cast<float*>(&Hbuf[0][0][0]);   // 16 KB used

    const int tid  = threadIdx.x;
    const int w    = tid >> 5;          // wave id 0..7
    const int lane = tid & 31;
    const int g    = lane >> 4;         // lane group
    const int n16  = lane & 15;         // column within 16-wide block

    // ---- persistent A fragments (held in VGPRs across all iterations) ----
    const h16* A1base = wsA;
    const h16* A2base = wsA + 16  * 512;
    const h16* A3base = wsA + 144 * 512;

    v16h A1[2], A2[2][8], A3;
#pragma unroll
    for (int rb = 0; rb < 2; ++rb) {
        A1[rb] = *(const v16h*)(A1base + (2 * w + rb) * 512 + lane * 16);
#pragma unroll
        for (int t = 0; t < 8; ++t)
            A2[rb][t] = *(const v16h*)(A2base + ((2 * w + rb) * 8 + t) * 512 + lane * 16);
    }
    A3 = *(const v16h*)(A3base + w * 512 + lane * 16);

    // ---- bias registers for this lane's D-layout slots ----
    float b1r[2][8], b2r[2][8];
#pragma unroll
    for (int rb = 0; rb < 2; ++rb)
#pragma unroll
        for (int r = 0; r < 8; ++r) {
            int M = (2 * w + rb) * 16 + r + 8 * g;
            b1r[rb][r] = b1[M];
            b2r[rb][r] = b2[M];
        }
    const int   Mred = w + 8 * g;                    // reduction-slot row (r==w)
    const float b4v  = (Mred < 3) ? b4[Mred] : 0.0f;

    for (int it = 0; it < ITERS; ++it) {
        const int colbase = (blockIdx.x * ITERS + it) * NT;

        // ================= GEMM1: h1 = sin(30*(W1 @ x + b1)) ==============
        v16h B1[2];
#pragma unroll
        for (int cb = 0; cb < 2; ++cb) {
            v16h b;
#pragma unroll
            for (int i = 0; i < 16; ++i) b[i] = (h16)0.0f;
            if (g == 0) {                            // group 0 holds K=0..15
                int pos = colbase + cb * 16 + n16;
                const float* srcs[4] = {s, nrm, pix, vdir};
#pragma unroll
                for (int h = 0; h < 12; ++h)
                    b[h] = (h16)srcs[h / 3][(h % 3) * NPOS + pos];
            }
            B1[cb] = b;                              // group 1 = K 16..31 = 0
        }

#pragma unroll
        for (int rb = 0; rb < 2; ++rb)
#pragma unroll
            for (int cb = 0; cb < 2; ++cb) {
                v8f c = {0.f, 0.f, 0.f, 0.f, 0.f, 0.f, 0.f, 0.f};
                c = __builtin_amdgcn_wmma_f32_16x16x32_f16(
                        false, A1[rb], false, B1[cb], (short)0, c, false, false);
                int col = cb * 16 + n16;
                int Kb  = (2 * w + rb) * 16 + 8 * g;
#pragma unroll
                for (int r = 0; r < 8; r += 2) {
                    union { h16 h[2]; unsigned u; } pk;
                    pk.h[0] = (h16)__sinf(OMEGA * (c[r]     + b1r[rb][r]));
                    pk.h[1] = (h16)__sinf(OMEGA * (c[r + 1] + b1r[rb][r + 1]));
                    *(unsigned*)&Hbuf[0][col][Kb + r] = pk.u;
                }
            }
        __syncthreads();

        // ================= GEMM2: h2 = sin(30*(W2 @ h1 + b2)) =============
        v8f acc[2][2];
#pragma unroll
        for (int rb = 0; rb < 2; ++rb)
#pragma unroll
            for (int cb = 0; cb < 2; ++cb)
                acc[rb][cb] = (v8f){0.f, 0.f, 0.f, 0.f, 0.f, 0.f, 0.f, 0.f};

#pragma unroll
        for (int t = 0; t < 8; ++t) {
            v16h Bt[2];
#pragma unroll
            for (int cb = 0; cb < 2; ++cb) {
                const h16* p = &Hbuf[0][cb * 16 + n16][t * 32 + 16 * g];
                v8h lo = *(const v8h*)p;
                v8h hi = *(const v8h*)(p + 8);
                Bt[cb] = __builtin_shufflevector(lo, hi, 0, 1, 2, 3, 4, 5, 6, 7,
                                                 8, 9, 10, 11, 12, 13, 14, 15);
            }
#pragma unroll
            for (int rb = 0; rb < 2; ++rb)
#pragma unroll
                for (int cb = 0; cb < 2; ++cb)
                    acc[rb][cb] = __builtin_amdgcn_wmma_f32_16x16x32_f16(
                        false, A2[rb][t], false, Bt[cb], (short)0, acc[rb][cb],
                        false, false);
        }

#pragma unroll
        for (int rb = 0; rb < 2; ++rb)
#pragma unroll
            for (int cb = 0; cb < 2; ++cb) {
                int col = cb * 16 + n16;
                int Kb  = (2 * w + rb) * 16 + 8 * g;
#pragma unroll
                for (int r = 0; r < 8; r += 2) {
                    union { h16 h[2]; unsigned u; } pk;
                    pk.h[0] = (h16)__sinf(OMEGA * (acc[rb][cb][r]     + b2r[rb][r]));
                    pk.h[1] = (h16)__sinf(OMEGA * (acc[rb][cb][r + 1] + b2r[rb][r + 1]));
                    *(unsigned*)&Hbuf[1][col][Kb + r] = pk.u;
                }
            }
        __syncthreads();   // all h1 reads done -> Hbuf[0] reusable as partials

        // ====== GEMM3 partials: wave w handles its own K slice (t = w) =====
#pragma unroll
        for (int cb = 0; cb < 2; ++cb) {
            const h16* p = &Hbuf[1][cb * 16 + n16][w * 32 + 16 * g];
            v8h lo = *(const v8h*)p;
            v8h hi = *(const v8h*)(p + 8);
            v16h Bt = __builtin_shufflevector(lo, hi, 0, 1, 2, 3, 4, 5, 6, 7,
                                              8, 9, 10, 11, 12, 13, 14, 15);
            v8f c = {0.f, 0.f, 0.f, 0.f, 0.f, 0.f, 0.f, 0.f};
            c = __builtin_amdgcn_wmma_f32_16x16x32_f16(
                    false, A3, false, Bt, (short)0, c, false, false);
#pragma unroll
            for (int r = 0; r < 8; ++r)
                part[(cb * 8 + w) * 256 + r * 32 + lane] = c[r];
        }
        __syncthreads();

        // ====== reduce partials + epilogue: out = z + 0.5*sin(30*(.+b4)) ===
        {
            const int r = tid >> 5;    // slot row == wave id
#pragma unroll
            for (int cb = 0; cb < 2; ++cb) {
                float sum = 0.0f;
#pragma unroll
                for (int ww = 0; ww < 8; ++ww)
                    sum += part[(cb * 8 + ww) * 256 + r * 32 + lane];
                if (Mred < 3) {
                    int pos = colbase + cb * 16 + n16;
                    out[Mred * NPOS + pos] =
                        z[Mred * NPOS + pos] + 0.5f * __sinf(OMEGA * (sum + b4v));
                }
            }
        }
        __syncthreads();   // before next iteration rewrites Hbuf[0]
    }
}

// ---------------------------------------------------------------------------
// Inputs (setup_inputs order): 0=z 1=n 2=ad(unused) 3=s 4=pixel 5=view
//                              6=W1 7=b1 8=W2 9=b2 10=W4 11=b4
// ---------------------------------------------------------------------------
extern "C" void kernel_launch(void* const* d_in, const int* in_sizes, int n_in,
                              void* d_out, int out_size, void* d_ws, size_t ws_size,
                              hipStream_t stream) {
    const float* z    = (const float*)d_in[0];
    const float* nrm  = (const float*)d_in[1];
    const float* s    = (const float*)d_in[3];
    const float* pix  = (const float*)d_in[4];
    const float* vdir = (const float*)d_in[5];
    const float* W1   = (const float*)d_in[6];
    const float* b1   = (const float*)d_in[7];
    const float* W2   = (const float*)d_in[8];
    const float* b2   = (const float*)d_in[9];
    const float* W4   = (const float*)d_in[10];
    const float* b4   = (const float*)d_in[11];
    h16* wsA = (h16*)d_ws;   // 152 fragments * 1024 B = 155648 B

    mlp_prep<<<304, 256, 0, stream>>>(W1, W2, W4, wsA);
    mlp_main<<<NWG, 256, 0, stream>>>(z, nrm, s, pix, vdir, wsA,
                                      b1, b2, b4, (float*)d_out);
}
